// DecoderRNN_59382217834730
// MI455X (gfx1250) — matrix operand order repaired and checked by reading, not compiled, not run on hardware
//
#include <hip/hip_runtime.h>
#include <hip/hip_bf16.h>

// ---------------------------------------------------------------------------
// DecoderRNN (attention LSTM decoder) for MI455X / gfx1250, wave32 + WMMA.
// GEMMs use v_wmma_f32_16x16x32_bf16 (bf16 inputs, fp32 accumulate); each
// wave owns TWO 16x16 N-tiles so the A-fragment is reused and two
// independent WMMA chains interleave. Softmax / LSTM / BatchNorm in fp32;
// LSTM pointwise fused into the BN1 reduction kernel.
// ---------------------------------------------------------------------------

#define kB 128
#define kT 20
#define kR 49
#define kD 2048
#define kE 512
#define kH 512
#define kV 10000
#define kXW (kE + kD)   // 2560, LSTM input width

typedef __attribute__((ext_vector_type(16))) __bf16 v16bf;
typedef __attribute__((ext_vector_type(8)))  __bf16 v8bf;
typedef __attribute__((ext_vector_type(8)))  float  v8f;

__device__ __forceinline__ unsigned short f2bfu(float f) {
  // round-to-nearest-even fp32 -> bf16
  unsigned u = __builtin_bit_cast(unsigned, f);
  return (unsigned short)((u + 0x7FFFu + ((u >> 16) & 1u)) >> 16);
}
__device__ __forceinline__ float sigmoidf(float x) {
  return 1.0f / (1.0f + __expf(-x));
}

// ---------------------------------------------------------------------------
// WMMA GEMM:  C[M,N] = (beta ? C : 0) + bias[N] + A[M,K](bf16) @ W[N,K](bf16)^T
// One wave -> two adjacent 16x16 N-tiles (A-fragment reused, 2 indep. WMMA
// chains); 4 waves per block along N => 8 N-tiles (128 cols) per block.
// A-fragment (16-bit A 16x32, ISA 7.12.2):
//   lanes 0-15 : M=lane,  elems 0..7 = K[k..k+7],    elems 8..15 = K[k+16..k+23]
//   lanes16-31 : M=ln-16, elems 0..7 = K[k+8..k+15], elems 8..15 = K[k+24..k+31]
// B-fragment: lane n = row n of W, same K order as A (hi*8 halves at +0,+16)
// C/D: VGPR v holds row (v + 8*(lane>=16)), col = lane&15.
// ---------------------------------------------------------------------------
__global__ void wmma_gemm_bf16(const __bf16* __restrict__ A, int lda,
                               const __bf16* __restrict__ W, int ldb,
                               const float* __restrict__ bias,
                               float* __restrict__ C, int ldc,
                               int N, int K, int beta, int relu) {
  const int wave = threadIdx.x >> 5;
  const int lane = threadIdx.x & 31;
  const int nt0  = (blockIdx.y * 4 + wave) * 2;   // first of two N-tiles
  if (nt0 * 16 >= N) return;                       // wave-uniform early-out
  const bool two = ((nt0 + 1) * 16) < N;           // second tile valid?
  const int m0 = blockIdx.x * 16;
  const int mr = lane & 15;
  const int hi = lane >> 4;
  const int col0 = nt0 * 16 + mr;
  const int col1 = two ? (col0 + 16) : col0;       // clamp when single tile

  v8f acc0, acc1;
  const float bv0 = bias ? bias[col0] : 0.0f;
  const float bv1 = bias ? bias[col1] : 0.0f;
#pragma unroll
  for (int v = 0; v < 8; ++v) {
    const int row = m0 + v + 8 * hi;
    acc0[v] = bv0 + (beta ? C[(size_t)row * ldc + col0] : 0.0f);
    acc1[v] = bv1 + ((beta && two) ? C[(size_t)row * ldc + col1] : 0.0f);
  }

  const __bf16* arow  = A + (size_t)(m0 + mr) * lda + hi * 8;
  const __bf16* brow0 = W + (size_t)col0 * ldb + hi * 8;
  const __bf16* brow1 = W + (size_t)col1 * ldb + hi * 8;  // ==brow0 if !two

  for (int k = 0; k < K; k += 32) {
    union { v16bf v; v8bf h[2]; } af;
    af.h[0] = *(const v8bf*)(arow + k);        // K[k + hi*8 .. +7]
    af.h[1] = *(const v8bf*)(arow + k + 16);   // K[k+16 + hi*8 .. +7]
    union { v16bf v; v8bf h[2]; } b0u, b1u;
    b0u.h[0] = *(const v8bf*)(brow0 + k);      // K[k + hi*8 .. +7]
    b0u.h[1] = *(const v8bf*)(brow0 + k + 16); // K[k+16 + hi*8 .. +7]
    b1u.h[0] = *(const v8bf*)(brow1 + k);
    b1u.h[1] = *(const v8bf*)(brow1 + k + 16);
    v16bf bf0 = b0u.v;
    v16bf bf1 = b1u.v;
    acc0 = __builtin_amdgcn_wmma_f32_16x16x32_bf16(
        false, af.v, false, bf0, (short)0, acc0, false, false);
    acc1 = __builtin_amdgcn_wmma_f32_16x16x32_bf16(
        false, af.v, false, bf1, (short)0, acc1, false, false);
  }

#pragma unroll
  for (int v = 0; v < 8; ++v) {
    const int row = m0 + v + 8 * hi;
    float x0 = acc0[v];
    if (relu) x0 = fmaxf(x0, 0.0f);
    C[(size_t)row * ldc + col0] = x0;
    if (two) {
      float x1 = acc1[v];
      if (relu) x1 = fmaxf(x1, 0.0f);
      C[(size_t)row * ldc + col1] = x1;
    }
  }
}

// ---------------------------------------------------------------------------
// Elementwise fp32 -> bf16 convert
// ---------------------------------------------------------------------------
__global__ void cvt_bf16_kernel(const float* __restrict__ src,
                                unsigned short* __restrict__ dst, int n) {
  int i = blockIdx.x * blockDim.x + threadIdx.x;
  if (i < n) dst[i] = f2bfu(src[i]);
}

// mean over R axis: features[B,R,D] -> meanf_bf[B,D] (bf16)
__global__ void mean_kernel(const float* __restrict__ feats,
                            unsigned short* __restrict__ meanf_bf) {
  int idx = blockIdx.x * blockDim.x + threadIdx.x;
  if (idx >= kB * kD) return;
  int b = idx / kD, d = idx % kD;
  float s = 0.0f;
  for (int r = 0; r < kR; ++r) s += feats[((size_t)b * kR + r) * kD + d];
  meanf_bf[(size_t)b * kD + d] = f2bfu(s * (1.0f / kR));
}

// embedding gather for step t: x_bf[b, 0:E] = bf16(emb[captions[b,t], :])
__global__ void embed_kernel(const int* __restrict__ cap,
                             const float* __restrict__ emb,
                             unsigned short* __restrict__ x_bf, int t) {
  int idx = blockIdx.x * blockDim.x + threadIdx.x;
  if (idx >= kB * kE) return;
  int b = idx / kE, e = idx % kE;
  int tok = cap[b * kT + t];
  x_bf[(size_t)b * kXW + e] = f2bfu(emb[(size_t)tok * kE + e]);
}

// attention: score = tanh(att1[b,r,:] + a2[b,:]) . va + bv ; softmax over r
__global__ void attn_kernel(const float* __restrict__ att1,
                            const float* __restrict__ a2,
                            const float* __restrict__ va,
                            const float* __restrict__ bv,
                            float* __restrict__ wbuf,
                            float* __restrict__ atten_out, int t) {
  __shared__ float sc[kR];
  __shared__ float ssum;
  const int b = blockIdx.x;
  const int lane = threadIdx.x & 31, wv = threadIdx.x >> 5;
  const float* a2r = a2 + (size_t)b * kH;
  for (int r = wv; r < kR; r += 8) {
    const float* row = att1 + ((size_t)b * kR + r) * kH;
    float s = 0.0f;
    for (int h = lane; h < kH; h += 32) s += tanhf(row[h] + a2r[h]) * va[h];
    for (int off = 16; off; off >>= 1) s += __shfl_xor(s, off, 32);
    if (lane == 0) sc[r] = s + bv[0];
  }
  __syncthreads();
  if (threadIdx.x == 0) {
    float m = -3.4e38f;
    for (int r = 0; r < kR; ++r) m = fmaxf(m, sc[r]);
    float s = 0.0f;
    for (int r = 0; r < kR; ++r) { sc[r] = __expf(sc[r] - m); s += sc[r]; }
    ssum = s;
  }
  __syncthreads();
  for (int r = threadIdx.x; r < kR; r += blockDim.x) {
    float wr = sc[r] / ssum;
    wbuf[b * kR + r] = wr;
    atten_out[((size_t)b * kT + t) * kR + r] = wr;
  }
}

// context: x_bf[b, E+d] = bf16( sum_r w[b,r] * features[b,r,d] )
__global__ void ctx_kernel(const float* __restrict__ wbuf,
                           const float* __restrict__ feats,
                           unsigned short* __restrict__ x_bf) {
  int idx = blockIdx.x * blockDim.x + threadIdx.x;
  if (idx >= kB * kD) return;
  int b = idx / kD, d = idx % kD;
  float s = 0.0f;
  for (int r = 0; r < kR; ++r)
    s += wbuf[b * kR + r] * feats[((size_t)b * kR + r) * kD + d];
  x_bf[(size_t)b * kXW + kE + d] = f2bfu(s);
}

// Fused LSTM pointwise (torch gate order i,f,g,o) + BatchNorm1d over batch.
// grid = kH columns, block = kB (=128) threads; thread tid = batch row.
// Updates c in place, writes post-BN h as bf16 (the carried state).
__global__ void lstm_bn_kernel(const float* __restrict__ gates,
                               float* __restrict__ c,
                               const float* __restrict__ gamma,
                               const float* __restrict__ beta_p,
                               unsigned short* __restrict__ h_bf) {
  __shared__ float red[kB];
  const int col = blockIdx.x;      // h index
  const int tid = threadIdx.x;     // batch index
  const float* g = gates + (size_t)tid * 4 * kH;
  float ig = sigmoidf(g[col]);
  float fg = sigmoidf(g[kH + col]);
  float gg = tanhf(g[2 * kH + col]);
  float og = sigmoidf(g[3 * kH + col]);
  float cn = fg * c[(size_t)tid * kH + col] + ig * gg;
  c[(size_t)tid * kH + col] = cn;
  float v = og * tanhf(cn);        // pre-BN hidden

  red[tid] = v; __syncthreads();
  for (int s = 64; s > 0; s >>= 1) { if (tid < s) red[tid] += red[tid + s]; __syncthreads(); }
  float mean = red[0] / (float)kB;
  __syncthreads();
  float d = v - mean;
  red[tid] = d * d; __syncthreads();
  for (int s = 64; s > 0; s >>= 1) { if (tid < s) red[tid] += red[tid + s]; __syncthreads(); }
  float var = red[0] / (float)kB;
  float y = d * rsqrtf(var + 1e-5f) * gamma[col] + beta_p[col];
  h_bf[(size_t)tid * kH + col] = f2bfu(y);
}

// BatchNorm over batch axis (biased stats), output bf16. grid = N cols,
// block = 128 (= B) threads.
__global__ void bn_kernel(const float* __restrict__ X,
                          const float* __restrict__ gamma,
                          const float* __restrict__ beta_p,
                          unsigned short* __restrict__ Ybf, int N) {
  __shared__ float red[kB];
  const int col = blockIdx.x;
  const int tid = threadIdx.x;
  float v = X[(size_t)tid * N + col];
  red[tid] = v; __syncthreads();
  for (int s = 64; s > 0; s >>= 1) { if (tid < s) red[tid] += red[tid + s]; __syncthreads(); }
  float mean = red[0] / (float)kB;
  __syncthreads();
  float d = v - mean;
  red[tid] = d * d; __syncthreads();
  for (int s = 64; s > 0; s >>= 1) { if (tid < s) red[tid] += red[tid + s]; __syncthreads(); }
  float var = red[0] / (float)kB;
  float y = d * rsqrtf(var + 1e-5f) * gamma[col] + beta_p[col];
  Ybf[(size_t)tid * N + col] = f2bfu(y);
}

// ---------------------------------------------------------------------------
// Host side
// ---------------------------------------------------------------------------
static inline void* ws_take(char*& p, size_t bytes) {
  void* r = (void*)p;
  p += (bytes + 255) & ~(size_t)255;
  return r;
}

static inline void gemm(hipStream_t s, const unsigned short* A, int lda,
                        const unsigned short* W, int ldb, const float* bias,
                        float* C, int ldc, int M, int N, int K,
                        int beta, int relu) {
  dim3 grid(M / 16, (N / 16 + 7) / 8);   // 4 waves x 2 N-tiles per block
  wmma_gemm_bf16<<<grid, 128, 0, s>>>((const __bf16*)A, lda, (const __bf16*)W,
                                      ldb, bias, C, ldc, N, K, beta, relu);
}

static inline void cvt(hipStream_t s, const float* src, unsigned short* dst, int n) {
  cvt_bf16_kernel<<<(n + 255) / 256, 256, 0, s>>>(src, dst, n);
}

extern "C" void kernel_launch(void* const* d_in, const int* in_sizes, int n_in,
                              void* d_out, int out_size, void* d_ws, size_t ws_size,
                              hipStream_t stream) {
  // inputs (setup_inputs order)
  const int*   captions = (const int*)  d_in[0];
  const float* features = (const float*)d_in[1];
  const float* emb      = (const float*)d_in[2];
  const float* Wa       = (const float*)d_in[3];
  const float* ba       = (const float*)d_in[4];
  const float* Ua       = (const float*)d_in[5];
  const float* bu       = (const float*)d_in[6];
  const float* va       = (const float*)d_in[7];
  const float* bv       = (const float*)d_in[8];
  const float* W_ih     = (const float*)d_in[9];
  const float* b_ih     = (const float*)d_in[10];
  const float* W_hh     = (const float*)d_in[11];
  const float* b_hh     = (const float*)d_in[12];
  const float* g1       = (const float*)d_in[13];
  const float* be1      = (const float*)d_in[14];
  const float* fc_w     = (const float*)d_in[15];
  const float* fc_b     = (const float*)d_in[16];
  const float* g2       = (const float*)d_in[17];
  const float* be2      = (const float*)d_in[18];
  const float* fc2_w    = (const float*)d_in[19];
  const float* fc2_b    = (const float*)d_in[20];
  const float* ih_w     = (const float*)d_in[21];
  const float* ih_b     = (const float*)d_in[22];
  const float* ic_w     = (const float*)d_in[23];
  const float* ic_b     = (const float*)d_in[24];

  float* outputs = (float*)d_out;                       // [B,T,V]
  float* atten   = outputs + (size_t)kB * kT * kV;      // [B,T,R]

  // workspace layout (~67 MB, 256B-aligned slices)
  char* p = (char*)d_ws;
  unsigned short* feat_bf  = (unsigned short*)ws_take(p, (size_t)kB*kR*kD*2);
  unsigned short* Wa_bf    = (unsigned short*)ws_take(p, (size_t)kH*kD*2);
  unsigned short* Ua_bf    = (unsigned short*)ws_take(p, (size_t)kH*kH*2);
  unsigned short* Wih_bf   = (unsigned short*)ws_take(p, (size_t)4*kH*kXW*2);
  unsigned short* Whh_bf   = (unsigned short*)ws_take(p, (size_t)4*kH*kH*2);
  unsigned short* fcw_bf   = (unsigned short*)ws_take(p, (size_t)256*kH*2);
  unsigned short* fc2w_bf  = (unsigned short*)ws_take(p, (size_t)kV*256*2);
  unsigned short* ihw_bf   = (unsigned short*)ws_take(p, (size_t)kH*kD*2);
  unsigned short* icw_bf   = (unsigned short*)ws_take(p, (size_t)kH*kD*2);
  unsigned short* meanf_bf = (unsigned short*)ws_take(p, (size_t)kB*kD*2);
  float* att1f  = (float*)ws_take(p, (size_t)kB*kR*kH*4);
  float* a2f    = (float*)ws_take(p, (size_t)kB*kH*4);
  float* wbuf   = (float*)ws_take(p, (size_t)kB*kR*4);
  unsigned short* x_bf = (unsigned short*)ws_take(p, (size_t)kB*kXW*2);
  float* gatesf = (float*)ws_take(p, (size_t)kB*4*kH*4);
  float* cbuf   = (float*)ws_take(p, (size_t)kB*kH*4);
  float* h0f    = (float*)ws_take(p, (size_t)kB*kH*4);
  unsigned short* h_bf = (unsigned short*)ws_take(p, (size_t)kB*kH*2);
  float* out1f  = (float*)ws_take(p, (size_t)kB*256*4);
  unsigned short* out1_bf = (unsigned short*)ws_take(p, (size_t)kB*256*2);

  // ---- one-time precompute ----
  cvt(stream, features, feat_bf, kB*kR*kD);
  cvt(stream, Wa,    Wa_bf,   kH*kD);
  cvt(stream, Ua,    Ua_bf,   kH*kH);
  cvt(stream, W_ih,  Wih_bf,  4*kH*kXW);
  cvt(stream, W_hh,  Whh_bf,  4*kH*kH);
  cvt(stream, fc_w,  fcw_bf,  256*kH);
  cvt(stream, fc2_w, fc2w_bf, kV*256);
  cvt(stream, ih_w,  ihw_bf,  kH*kD);
  cvt(stream, ic_w,  icw_bf,  kH*kD);
  mean_kernel<<<(kB*kD + 255)/256, 256, 0, stream>>>(features, meanf_bf);

  // h0 = mean_f @ ih_w^T + ih_b ; c0 = mean_f @ ic_w^T + ic_b
  gemm(stream, meanf_bf, kD, ihw_bf, kD, ih_b, h0f,  kH, kB, kH, kD, 0, 0);
  cvt(stream, h0f, h_bf, kB*kH);
  gemm(stream, meanf_bf, kD, icw_bf, kD, ic_b, cbuf, kH, kB, kH, kD, 0, 0);

  // att1 = features @ Wa^T + ba : [B*R, H]
  gemm(stream, feat_bf, kD, Wa_bf, kD, ba, att1f, kH, kB*kR, kH, kD, 0, 0);

  // ---- recurrence ----
  for (int t = 0; t < kT; ++t) {
    // a2 = h @ Ua^T + bu
    gemm(stream, h_bf, kH, Ua_bf, kH, bu, a2f, kH, kB, kH, kH, 0, 0);
    // attention scores + softmax (also writes atten_weights output)
    attn_kernel<<<kB, 256, 0, stream>>>(att1f, a2f, va, bv, wbuf, atten, t);
    // ctx -> x_bf[:, E:E+D]; embedding -> x_bf[:, 0:E]
    ctx_kernel<<<(kB*kD + 255)/256, 256, 0, stream>>>(wbuf, features, x_bf);
    embed_kernel<<<(kB*kE + 255)/256, 256, 0, stream>>>(captions, emb, x_bf, t);
    // gates = x @ W_ih^T + b_ih + h @ W_hh^T + b_hh
    gemm(stream, x_bf, kXW, Wih_bf, kXW, b_ih, gatesf, 4*kH, kB, 4*kH, kXW, 0, 0);
    gemm(stream, h_bf, kH,  Whh_bf, kH,  b_hh, gatesf, 4*kH, kB, 4*kH, kH,  1, 0);
    // fused LSTM pointwise + BN1 -> carried h (bf16), c updated in place
    lstm_bn_kernel<<<kH, kB, 0, stream>>>(gatesf, cbuf, g1, be1, h_bf);
    // out1 = relu(h @ fc_w^T + fc_b); out1 = BN2(out1)
    gemm(stream, h_bf, kH, fcw_bf, kH, fc_b, out1f, 256, kB, 256, kH, 0, 1);
    bn_kernel<<<256, kB, 0, stream>>>(out1f, g2, be2, out1_bf, 256);
    // outputs[:, t, :] = out1 @ fc2_w^T + fc2_b   (ldc = T*V strided store)
    gemm(stream, out1_bf, 256, fc2w_bf, 256, fc2_b,
         outputs + (size_t)t * kV, kT * kV, kB, kV, 256, 0, 0);
  }
  (void)in_sizes; (void)n_in; (void)out_size; (void)ws_size;
}
